// alpha_net_spatial_14843406975139
// MI455X (gfx1250) — matrix operand order repaired and checked by reading, not compiled, not run on hardware
//
#include <hip/hip_runtime.h>

typedef __attribute__((ext_vector_type(16))) _Float16 v16h;
typedef __attribute__((ext_vector_type(8)))  _Float16 v8h;
typedef __attribute__((ext_vector_type(8)))  float    v8f;

#define NBATCH 32
#define NCH    256
#define NTOK   1024
#define KCH    32          // WMMA K per chunk
#define NCHUNK (NTOK / KCH)
#define LSTR   40          // f16 row stride: 80B -> rows 16B-aligned, 20-bank
                           // stride (distinct banks over 16 consecutive rows)

union H2 { _Float16 f[2]; unsigned int u; };

// CDNA5 has a hardware TRANS tanh (V_TANH_F32); use the builtin when the
// toolchain exposes it, otherwise fall back to libm (compiler handles the
// TRANS hazard rules for the builtin).
__device__ __forceinline__ float fast_tanh(float x)
{
#if defined(__has_builtin)
#if __has_builtin(__builtin_amdgcn_tanhf)
    return __builtin_amdgcn_tanhf(x);
#else
    return tanhf(x);
#endif
#else
    return tanhf(x);
#endif
}

// ---------------------------------------------------------------------------
// Kernel 1: scalar projections a[b,n], k[b,n] (biases folded in).
// ---------------------------------------------------------------------------
__global__ __launch_bounds__(256)
void proj_kernel(const float* __restrict__ h,
                 const float* __restrict__ w1, const float* __restrict__ b1,
                 const float* __restrict__ w2, const float* __restrict__ b2,
                 float* __restrict__ abuf, float* __restrict__ kbuf)
{
    const int b = blockIdx.y;
    const int n = blockIdx.x * 256 + threadIdx.x;
    const float* base = h + (size_t)b * NCH * NTOK + n;
    float sa = 0.f, sk = 0.f;
    #pragma unroll 4
    for (int c = 0; c < NCH; ++c) {
        const float x = base[(size_t)c * NTOK];
        sa = fmaf(x, w1[c], sa);
        sk = fmaf(x, w2[c], sk);
    }
    abuf[b * NTOK + n] = sa + b1[0];
    kbuf[b * NTOK + n] = sk + b2[0];
}

// ---------------------------------------------------------------------------
// Kernel 2: fused attention,  D[c,i] = sum_j h[b,c,j] * exp(tanh(a_i + k_j))
//           out[b,c,i] = leaky( D[c,i] / rowsum_i )
//
// Block: one batch, 128 i-columns, all 256 channels. 8 waves arranged as
// (cg in {0,1}) x (ig in {0..3}): wave owns 8 c-tiles and 2 i-tiles (32 i's).
// Each A fragment (X^T tile from LDS) feeds two WMMAs (B fragments for the
// two i-tiles), halving LDS read bytes per WMMA. LDS is double-buffered so
// staging chunk n+1 overlaps compute of chunk n (one barrier per chunk).
// ---------------------------------------------------------------------------
__global__ __launch_bounds__(256)
void attn_kernel(const float* __restrict__ h,
                 const float* __restrict__ abuf,
                 const float* __restrict__ kbuf,
                 float* __restrict__ out)
{
    __shared__ __align__(16) _Float16 Xlds[2][NCH * LSTR];  // 2 x 20.0 KB
    __shared__ __align__(16) float kchs[2][KCH];

    const int b    = blockIdx.x;
    const int it   = blockIdx.y;
    const int t    = threadIdx.x;
    const int wave = t >> 5;
    const int lane = t & 31;
    const int sub  = lane >> 4;          // half-wave select
    const int l16  = lane & 15;
    const int ig   = wave & 3;           // i-group (4 x 32 i's = 128)
    const int cg   = wave >> 2;          // c-group (2 x 8 c-tiles = 256 ch)

    const int iA = it * 128 + ig * 32 + l16;   // first i-tile column
    const int iB = iA + 16;                    // second i-tile column

    const float  aA = abuf[b * NTOK + iA];
    const float  aB = abuf[b * NTOK + iB];
    const float* kb = kbuf + b * NTOK;
    const float* hB = h + (size_t)b * NCH * NTOK;

    v8f acc[16];                          // [0..7]: i-tile A, [8..15]: i-tile B
    #pragma unroll
    for (int q = 0; q < 16; ++q) {
        #pragma unroll
        for (int e = 0; e < 8; ++e) acc[q][e] = 0.f;
    }
    float rsA = 0.f, rsB = 0.f;

    const int sc = t >> 3;   // staging: channel within pass (0..31)
    const int sp = t & 7;    // staging: float4 slot (0..7)

    // stage h[b, :, j0..j0+32), f32 -> f16, into LDS buffer `buf`
    auto stage = [&](int buf, int j0) {
        #pragma unroll
        for (int pass = 0; pass < 8; ++pass) {
            const int c = sc + pass * 32;
            const float4 v = *(const float4*)(hB + (size_t)c * NTOK + j0 + sp * 4);
            H2 lo, hi;
            lo.f[0] = (_Float16)v.x; lo.f[1] = (_Float16)v.y;
            hi.f[0] = (_Float16)v.z; hi.f[1] = (_Float16)v.w;
            uint2 pk; pk.x = lo.u; pk.y = hi.u;
            *(uint2*)&Xlds[buf][c * LSTR + sp * 4] = pk;   // 8B-aligned
        }
        if (t < KCH) kchs[buf][t] = kb[j0 + t];
    };

    auto compute = [&](int buf) {
        // this half-wave's 16 k values: contiguous 64B -> 4x ds_load_b128
        const float4* kv4 = (const float4*)&kchs[buf][sub * 16];
        const float4 k4[4] = { kv4[0], kv4[1], kv4[2], kv4[3] };
        const float* kv = (const float*)&k4[0];

        // B fragments: P^T (32 x 16) for the wave's two i-tiles.
        // B k order used here (contiguous, same as A): VGPR v holds K = 16*sub + 2v, 2v+1; N = l16.
        v16h bfA, bfB;
        #pragma unroll
        for (int e = 0; e < 16; ++e) {
            const float kvv = kv[e];
            const float pA = __expf(fast_tanh(aA + kvv));
            const float pB = __expf(fast_tanh(aB + kvv));
            rsA += pA; rsB += pB;
            bfA[e] = (_Float16)pA;
            bfB[e] = (_Float16)pB;
        }
        // 8 channel tiles, each A fragment reused for both i-tiles.
        // A k order used here (contiguous, same as B): M = l16; K = 16*sub+e
        #pragma unroll
        for (int ctl = 0; ctl < 8; ++ctl) {
            const _Float16* row = &Xlds[buf][((cg * 8 + ctl) * 16 + l16) * LSTR];
            const v8h lo = *(const v8h*)(row + sub * 16);       // 16B-aligned
            const v8h hi = *(const v8h*)(row + sub * 16 + 8);   // 16B-aligned
            const v16h af = __builtin_shufflevector(
                lo, hi, 0,1,2,3,4,5,6,7,8,9,10,11,12,13,14,15);
            acc[ctl] = __builtin_amdgcn_wmma_f32_16x16x32_f16(
                false, af, false, bfA, (short)0, acc[ctl], false, false);
            acc[8 + ctl] = __builtin_amdgcn_wmma_f32_16x16x32_f16(
                false, af, false, bfB, (short)0, acc[8 + ctl], false, false);
        }
    };

    stage(0, 0);
    __syncthreads();
    #pragma unroll 1
    for (int chunk = 0; chunk < NCHUNK; ++chunk) {
        const int cur = chunk & 1;
        if (chunk + 1 < NCHUNK) stage(cur ^ 1, (chunk + 1) * KCH);
        compute(cur);
        __syncthreads();
    }

    // every D element this lane holds shares i (per half: iA / iB)
    rsA += __shfl_xor(rsA, 16, 32);
    rsB += __shfl_xor(rsB, 16, 32);
    const float invA = 1.0f / rsA;
    const float invB = 1.0f / rsB;

    // D layout: VGPR r -> M(channel in tile) = r + 8*sub; N(i) = l16.
    #pragma unroll
    for (int ctl = 0; ctl < 8; ++ctl) {
        #pragma unroll
        for (int r = 0; r < 8; ++r) {
            const int c = (cg * 8 + ctl) * 16 + sub * 8 + r;
            float vA = acc[ctl][r] * invA;
            float vB = acc[8 + ctl][r] * invB;
            vA = (vA >= 0.f) ? vA : 0.2f * vA;
            vB = (vB >= 0.f) ? vB : 0.2f * vB;
            out[((size_t)b * NCH + c) * NTOK + iA] = vA;
            out[((size_t)b * NCH + c) * NTOK + iB] = vB;
        }
    }
}

// ---------------------------------------------------------------------------
extern "C" void kernel_launch(void* const* d_in, const int* in_sizes, int n_in,
                              void* d_out, int out_size, void* d_ws, size_t ws_size,
                              hipStream_t stream)
{
    const float* h   = (const float*)d_in[0];
    const float* w1w = (const float*)d_in[1];
    const float* w1b = (const float*)d_in[2];
    const float* w2w = (const float*)d_in[3];
    const float* w2b = (const float*)d_in[4];
    float* out = (float*)d_out;

    float* abuf = (float*)d_ws;                 // [32*1024] f32
    float* kbuf = abuf + NBATCH * NTOK;         // [32*1024] f32  (256 KB total)

    proj_kernel<<<dim3(NTOK / 256, NBATCH), 256, 0, stream>>>(
        h, w1w, w1b, w2w, w2b, abuf, kbuf);
    attn_kernel<<<dim3(NBATCH, NTOK / 128), 256, 0, stream>>>(
        h, abuf, kbuf, out);
}
